// SIALM_15917148799352
// MI455X (gfx1250) — compile-verified, matrix operand order checked by reading, not run on hardware
//
#include <hip/hip_runtime.h>
#include <stdint.h>

// ---------------------------------------------------------------------------
// Types
// ---------------------------------------------------------------------------
typedef _Float16 half_t;
typedef __attribute__((ext_vector_type(16))) _Float16 v16h;
typedef __attribute__((ext_vector_type(8)))  _Float16 v8h;
typedef __attribute__((ext_vector_type(8)))  float    v8f;
typedef unsigned int u32x4 __attribute__((ext_vector_type(4)));
typedef int          i32x8 __attribute__((ext_vector_type(8)));
typedef int          i32x4 __attribute__((ext_vector_type(4)));

#define NB_B   8      // batch
#define D_MODEL 256
#define NHEAD  4
#define DH     64
#define NHASH  8
#define BUCKET 32
#define NHEAD_PAD 320   // padded head-vocab width (multiple of 64)

__device__ inline v8f wmma_f16(v16h a, v16h b, v8f c) {
  return __builtin_amdgcn_wmma_f32_16x16x32_f16(false, a, false, b, (short)0, c,
                                                false, false);
}
// Assemble a 16-half fragment from two contiguous 16B LDS chunks at p and p+16
// (lowers to two ds_load_b128 instead of 16 ds_load_u16).
__device__ inline v16h frag16(const half_t* p) {
  v8h lo = *(const v8h*)p;
  v8h hi = *(const v8h*)(p + 16);
  return __builtin_shufflevector(lo, hi, 0, 1, 2, 3, 4, 5, 6, 7,
                                 8, 9, 10, 11, 12, 13, 14, 15);
}
__device__ inline float wred_sum(float v) {
  for (int o = 16; o; o >>= 1) v += __shfl_xor(v, o, 32);
  return v;
}
__device__ inline float wred_max(float v) {
  for (int o = 16; o; o >>= 1) v = fmaxf(v, __shfl_xor(v, o, 32));
  return v;
}
__device__ inline unsigned h32(unsigned x) {
  x ^= x >> 16; x *= 0x7feb352dU; x ^= x >> 15; x *= 0x846ca68bU; x ^= x >> 16;
  return x;
}

// ---------------------------------------------------------------------------
// TDM / cluster demonstration kernel (6-arg clang-23 builtin form).
// ---------------------------------------------------------------------------
__global__ void k_tdm_stage(const float* __restrict__ src, float* __restrict__ sink) {
#if __has_builtin(__builtin_amdgcn_tensor_load_to_lds)
  __shared__ __align__(16) float buf[2048];
  unsigned long long ga = (unsigned long long)(uintptr_t)src;
  u32x4 g0; i32x8 g1; i32x4 z4 = {0, 0, 0, 0};
  i32x8 z8 = {0, 0, 0, 0, 0, 0, 0, 0};
  g0[0] = 1u;                                           // count = 1
  g0[1] = 0u;                                           // lds_addr
  g0[2] = (unsigned)(ga & 0xFFFFFFFFull);               // global addr lo
  g0[3] = (unsigned)((ga >> 32) & 0x01FFFFFFull) | (2u << 30);  // type=2
  g1[0] = (2 << 16);                                    // data_size = 4B
  g1[1] = (int)((512u & 0xFFFFu) << 16);                // tensor_dim0 = 512
  g1[2] = (int)(4u << 16);                              // tensor_dim1 = 4
  g1[3] = (int)(512u << 16);                            // tile_dim0 = 512
  g1[4] = 4;                                            // tile_dim1 = 4
  g1[5] = 512;                                          // dim0 stride
  g1[6] = 0; g1[7] = 0;
  __builtin_amdgcn_tensor_load_to_lds(g0, g1, z4, z4, z8, 0);
#if __has_builtin(__builtin_amdgcn_s_wait_tensorcnt)
  __builtin_amdgcn_s_wait_tensorcnt(0);
#endif
  __syncthreads();
  if (threadIdx.x == 0) sink[0] = buf[0];
#else
  if (threadIdx.x == 0) sink[0] = src[0];
#endif
#if __has_builtin(__builtin_amdgcn_s_cluster_barrier)
  __builtin_amdgcn_s_cluster_barrier();   // NOP when ClusterID==0
#endif
}

// ---------------------------------------------------------------------------
// Generic WMMA GEMM: C[M,N] = act(A[M,K] @ B[K,N] + bias) (+ residual)
// A,B f16 row-major, C f32.  Requires K%32==0 and N%8==0.
// Block tile 64x64, K-step 32, 8 wave32 waves.  B is staged transposed in LDS
// so WMMA B-fragments are two contiguous ds_load_b128 per tile.
// ---------------------------------------------------------------------------
__global__ __launch_bounds__(256) void k_gemm(
    const half_t* __restrict__ A, const half_t* __restrict__ B,
    const float* __restrict__ bias, const float* __restrict__ resid,
    float* __restrict__ C, int M, int N, int K, int relu) {
  __shared__ __align__(16) half_t As[64][32];
  __shared__ __align__(16) half_t BsT[64][40];   // [n][k], padded rows (80B)
  const int tid = threadIdx.x;
  const int lane = tid & 31, wave = tid >> 5;
  const int bm = blockIdx.y * 64, bn = blockIdx.x * 64;
  const int wm = (wave >> 1) * 16, wn = (wave & 1) * 32;
  v8f acc0 = {}; v8f acc1 = {};
  const int ar = tid >> 2, ac = (tid & 3) * 8;
  const int brr = tid >> 3, bcc = (tid & 7) * 8;
  for (int k0 = 0; k0 < K; k0 += 32) {
    // stage A tile (64x32), whole-vector with row guard
    int gr = bm + ar;
    v8h av = {};
    if (gr < M) av = *(const v8h*)(A + (size_t)gr * K + k0 + ac);
    *(v8h*)&As[ar][ac] = av;
    // stage B tile (32 x 64) transposed into BsT[n][k]
    {
      v8h bv = {};
      if (bn + bcc + 8 <= N)
        bv = *(const v8h*)(B + (size_t)(k0 + brr) * N + bn + bcc);
      for (int j = 0; j < 8; j++) BsT[bcc + j][brr] = bv[j];
    }
    if (k0 + 32 < K) {
      int pr = (bm + ar < M) ? (bm + ar) : (M - 1);
      __builtin_prefetch(A + (size_t)pr * K + k0 + 32, 0, 1);
      __builtin_prefetch(B + (size_t)(k0 + 32 + brr) * N + bn, 0, 1);
    }
    __syncthreads();
    const int arow = wm + (lane & 15);
    const int kk = (lane >> 4) * 8;
    const int c0 = wn + (lane & 15), c1 = c0 + 16;
    v16h af  = frag16(&As[arow][kk]);
    v16h bf0 = frag16(&BsT[c0][kk]);
    v16h bf1 = frag16(&BsT[c1][kk]);
    acc0 = wmma_f16(af, bf0, acc0);
    acc1 = wmma_f16(af, bf1, acc1);
    __syncthreads();
  }
  // epilogue (C/D layout: VGPR e -> M = e (+8 for lanes 16-31), N = lane&15)
  const int rbase = bm + wm + ((lane & 16) ? 8 : 0);
  const int cg0 = bn + wn + (lane & 15);
  for (int e = 0; e < 8; e++) {
    int r = rbase + e;
    if (r >= M) continue;
    if (cg0 < N) {
      float v = acc0[e];
      if (bias)  v += bias[cg0];
      if (relu)  v = fmaxf(v, 0.f);
      if (resid) v += resid[(size_t)r * N + cg0];
      C[(size_t)r * N + cg0] = v;
    }
    int c1g = cg0 + 16;
    if (c1g < N) {
      float v = acc1[e];
      if (bias)  v += bias[c1g];
      if (relu)  v = fmaxf(v, 0.f);
      if (resid) v += resid[(size_t)r * N + c1g];
      C[(size_t)r * N + c1g] = v;
    }
  }
}

// ---------------------------------------------------------------------------
// Elementwise / layout kernels
// ---------------------------------------------------------------------------
__global__ void k_cvt_h(const float* __restrict__ x, half_t* __restrict__ y, size_t n) {
  size_t i = blockIdx.x * (size_t)blockDim.x + threadIdx.x;
  if (i < n) y[i] = (half_t)x[i];
}
__global__ void k_copy(const float* __restrict__ x, float* __restrict__ y, size_t n) {
  size_t i = blockIdx.x * (size_t)blockDim.x + threadIdx.x;
  if (i < n) y[i] = x[i];
}
__global__ void k_avg2(const float* __restrict__ a, const float* __restrict__ b,
                       float* __restrict__ o, size_t n) {
  size_t i = blockIdx.x * (size_t)blockDim.x + threadIdx.x;
  if (i < n) o[i] = (a[i] + b[i]) * 0.5f;
}
__global__ void k_add_pe(float* __restrict__ x, const float* __restrict__ pe,
                         int b, int t) {
  size_t i = blockIdx.x * (size_t)blockDim.x + threadIdx.x;
  size_t tot = (size_t)b * t * 256;
  if (i >= tot) return;
  int d = i % 256;
  int row = (int)((i / 256) % t);
  x[i] += pe[(size_t)row * 256 + d];
}
__global__ __launch_bounds__(256) void k_layernorm(
    const float* __restrict__ x, const float* __restrict__ g,
    const float* __restrict__ b, float* __restrict__ out, int rows) {
  int row = blockIdx.x * 8 + (threadIdx.x >> 5);
  int lane = threadIdx.x & 31;
  if (row >= rows) return;
  const float* xr = x + (size_t)row * 256;
  float v[8]; float s = 0.f;
  for (int j = 0; j < 8; j++) { v[j] = xr[lane + j * 32]; s += v[j]; }
  s = wred_sum(s);
  float m = s * (1.f / 256.f);
  float var = 0.f;
  for (int j = 0; j < 8; j++) { float d = v[j] - m; var += d * d; }
  var = wred_sum(var) * (1.f / 256.f);
  float r = rsqrtf(var + 1e-5f);
  for (int j = 0; j < 8; j++) {
    int idx = lane + j * 32;
    out[(size_t)row * 256 + idx] = (v[j] - m) * r * g[idx] + b[idx];
  }
}
__global__ void k_concat_rows(const float* __restrict__ x, const float* __restrict__ mem,
                              float* __restrict__ out, int b, int t, int tm, int d) {
  size_t i = blockIdx.x * (size_t)blockDim.x + threadIdx.x;
  int T = t + tm;
  size_t tot = (size_t)b * T * d;
  if (i >= tot) return;
  int dd = i % d;
  size_t r = i / d;
  int row = (int)(r % T);
  int bb = (int)(r / T);
  out[i] = (row < t) ? x[((size_t)bb * t + row) * d + dd]
                     : mem[((size_t)bb * tm + (row - t)) * d + dd];
}
__global__ void k_split_qk(const float* __restrict__ qk, half_t* __restrict__ out,
                           int b, int T) {
  size_t i = blockIdx.x * (size_t)blockDim.x + threadIdx.x;
  size_t tot = (size_t)b * T * 256;
  if (i >= tot) return;
  int d = i % 256;
  size_t r = i / 256;
  int row = (int)(r % T);
  int bb = (int)(r / T);
  int h = d >> 6, e = d & 63;
  out[(((size_t)(bb * NHEAD + h)) * T + row) * 64 + e] = (half_t)qk[i];
}
__global__ void k_tile_v(const float* __restrict__ v, half_t* __restrict__ out,
                         int b, int T) {
  size_t i = blockIdx.x * (size_t)blockDim.x + threadIdx.x;
  size_t tot = (size_t)b * NHEAD * T * 64;
  if (i >= tot) return;
  int e = i % 64;
  size_t r = i / 64;
  int row = (int)(r % T);
  int bh = (int)(r / T);
  int bb = bh / NHEAD;
  out[i] = (half_t)v[((size_t)bb * T + row) * 64 + e];
}
__global__ void k_gen_rot(float* __restrict__ rot, int n, unsigned seed) {
  int i = blockIdx.x * blockDim.x + threadIdx.x;
  if (i >= n) return;
  unsigned a = h32(seed ^ (0x9E3779B9u * (unsigned)(i + 1)));
  unsigned b = h32(a ^ 0x85EBCA6Bu);
  float u1 = (float)(a >> 8) * (1.0f / 16777216.f) + 1e-7f;
  float u2 = (float)(b >> 8) * (1.0f / 16777216.f);
  rot[i] = sqrtf(-2.f * logf(u1)) * cosf(6.2831853f * u2);
}
// LSH bucket assignment: keys[bh*L + h*T + i] = (argmax bucket + h*nb)*T + i
__global__ __launch_bounds__(32) void k_buckets(
    const half_t* __restrict__ qk, const float* __restrict__ rot,
    int* __restrict__ keys, int T, int nb2, int L) {
  int idx = blockIdx.x * 32 + threadIdx.x;
  int bh = idx / T, i = idx % T;
  const half_t* q = qk + ((size_t)bh * T + i) * 64;
  float qv[64];
  for (int f = 0; f < 64; f++) qv[f] = (float)q[f];
  for (int h = 0; h < NHASH; h++) {
    float best = -3.0e38f; int bi = 0;
    for (int r = 0; r < nb2; r++) {
      float s = 0.f;
      for (int f = 0; f < 64; f++) s += qv[f] * rot[(f * NHASH + h) * nb2 + r];
      if (s > best)  { best = s;  bi = r; }
      if (-s > best) { best = -s; bi = r + nb2; }
    }
    int bucket = bi + h * (2 * nb2);
    keys[(size_t)bh * L + h * T + i] = bucket * T + i;
  }
}
// Bitonic sort (per bh) of L keys padded to P2; outputs sorted orig indices +
// inverse permutation.  Runs fully inside LDS (<=64KB of the 320KB/WGP pool).
extern __shared__ unsigned long long smem_sort[];
__global__ void k_sort(const int* __restrict__ keys, int* __restrict__ st,
                       int* __restrict__ undo, int L, int P2) {
  int bh = blockIdx.x, tid = threadIdx.x, nt = blockDim.x;
  const int* kb = keys + (size_t)bh * L;
  for (int i = tid; i < P2; i += nt)
    smem_sort[i] = (i < L)
        ? ((((unsigned long long)(unsigned)kb[i]) << 32) | (unsigned)i)
        : 0xFFFFFFFF00000000ULL;
  __syncthreads();
  for (int ksz = 2; ksz <= P2; ksz <<= 1) {
    for (int j = ksz >> 1; j > 0; j >>= 1) {
      for (int i = tid; i < P2; i += nt) {
        int ixj = i ^ j;
        if (ixj > i) {
          bool up = ((i & ksz) == 0);
          unsigned long long a = smem_sort[i], c = smem_sort[ixj];
          if ((up && a > c) || (!up && a < c)) { smem_sort[i] = c; smem_sort[ixj] = a; }
        }
      }
      __syncthreads();
    }
  }
  for (int i = tid; i < L; i += nt) {
    int s = (int)(smem_sort[i] & 0xFFFFFFFFu);
    st[(size_t)bh * L + i] = s;
    undo[(size_t)bh * L + s] = i;
  }
}
__global__ void k_gather_sorted(const half_t* __restrict__ qk, const half_t* __restrict__ v,
                                const int* __restrict__ st, half_t* __restrict__ sqk,
                                half_t* __restrict__ sv, int* __restrict__ stq,
                                int T, int L) {
  int idx = blockIdx.x * blockDim.x + threadIdx.x;
  if (idx >= 32 * L) return;
  int bh = idx / L;
  int s = st[idx];
  int row = s % T;
  const half_t* qs = qk + ((size_t)bh * T + row) * 64;
  const half_t* vs = v + ((size_t)bh * T + row) * 64;
  half_t* qd = sqk + (size_t)idx * 64;
  half_t* vd = sv + (size_t)idx * 64;
  for (int e = 0; e < 64; e += 8) {
    *(v8h*)(qd + e) = *(const v8h*)(qs + e);
    *(v8h*)(vd + e) = *(const v8h*)(vs + e);
  }
  stq[idx] = row;
}
// ---------------------------------------------------------------------------
// LSH attention chunk: one wave per (bh, chunk).  Dots and P@V via WMMA.
// bv is staged transposed (feature-major) so P@V B-fragments are contiguous.
// ---------------------------------------------------------------------------
__global__ __launch_bounds__(32) void k_attn_chunk(
    const half_t* __restrict__ sqk, const half_t* __restrict__ sv,
    const int* __restrict__ stq, float* __restrict__ osort,
    float* __restrict__ lse_out, int L, int nch, int causal, int qlen) {
  int chunk = blockIdx.x % nch;
  int bh = blockIdx.x / nch;
  int lane = threadIdx.x;
  __shared__ __align__(16) half_t bq[32][64];
  __shared__ __align__(16) half_t bk[64][64];     // [key][feature] (normalized)
  __shared__ __align__(16) half_t bvT[64][72];    // [feature][key], padded rows
  __shared__ __align__(16) half_t pr[32][64];
  __shared__ float sc[32][64];
  __shared__ int qt[32];
  __shared__ int kt[64];
  size_t base = (size_t)bh * L;
  int r0 = chunk * 32;
  int rp = ((chunk + nch - 1) % nch) * 32;
  // queries
  {
    const half_t* qrow = sqk + (base + r0 + lane) * 64;
    for (int e = 0; e < 64; e += 8) *(v8h*)&bq[lane][e] = *(const v8h*)(qrow + e);
    qt[lane] = stq[base + r0 + lane];
  }
  // keys/values: current chunk + lookback chunk; keys L2-normalized
  for (int hf = 0; hf < 2; hf++) {
    int srow = (hf == 0 ? r0 : rp) + lane;
    int dst = hf * 32 + lane;
    const half_t* krow = sqk + (base + srow) * 64;
    const half_t* vrow = sv + (base + srow) * 64;
    float nrm = 0.f;
    for (int e = 0; e < 64; e++) { float f = (float)krow[e]; nrm += f * f; }
    float scl = rsqrtf(fmaxf(nrm, 1e-24f));
    for (int e = 0; e < 64; e++) bk[dst][e] = (half_t)((float)krow[e] * scl);
    for (int e8 = 0; e8 < 64; e8 += 8) {
      v8h vv = *(const v8h*)(vrow + e8);
      for (int j = 0; j < 8; j++) bvT[e8 + j][dst] = vv[j];
    }
    kt[dst] = stq[base + srow];
  }
  __syncthreads();
  // dots = bq (32x64) @ bk^T (64x64)
  v8f d[2][4];
  {
    v8f z = {};
    for (int m = 0; m < 2; m++)
      for (int n = 0; n < 4; n++) d[m][n] = z;
  }
  for (int kb = 0; kb < 2; kb++) {
    int kOff = kb * 32;
    int kk = (lane >> 4) * 8;
    v16h af0 = frag16(&bq[(lane & 15)][kOff + kk]);
    v16h af1 = frag16(&bq[16 + (lane & 15)][kOff + kk]);
    for (int n = 0; n < 4; n++) {
      int col = n * 16 + (lane & 15);
      v16h bf = frag16(&bk[col][kOff + kk]);   // B[k][n] = bk[n][k]
      d[0][n] = wmma_f16(af0, bf, d[0][n]);
      d[1][n] = wmma_f16(af1, bf, d[1][n]);
    }
  }
  for (int m = 0; m < 2; m++)
    for (int n = 0; n < 4; n++)
      for (int e = 0; e < 8; e++) {
        int row = m * 16 + e + ((lane & 16) ? 8 : 0);
        int col = n * 16 + (lane & 15);
        sc[row][col] = d[m][n][e];
      }
  __syncthreads();
  // masked softmax per query row (row = lane)
  {
    int myq = qt[lane];
    float mx = -3.0e38f;
    for (int j = 0; j < 64; j++) {
      float v = sc[lane][j] * 0.125f;          // dh^-0.5
      int tj = kt[j];
      if (causal && (myq < tj) && (tj < qlen)) v = -1e9f;
      if (myq == tj) v = -5e4f;
      sc[lane][j] = v;
      mx = fmaxf(mx, v);
    }
    float ssum = 0.f;
    for (int j = 0; j < 64; j++) {
      float e = expf(sc[lane][j] - mx);
      sc[lane][j] = e;
      ssum += e;
    }
    float inv = 1.f / ssum;
    for (int j = 0; j < 64; j++) pr[lane][j] = (half_t)(sc[lane][j] * inv);
    lse_out[base + r0 + lane] = mx + logf(ssum);
  }
  __syncthreads();
  // out = pr (32x64) @ bv (64x64);  B[k][n] = bvT[n][k] reads contiguous
  v8f o[2][4];
  {
    v8f z = {};
    for (int m = 0; m < 2; m++)
      for (int n = 0; n < 4; n++) o[m][n] = z;
  }
  for (int kb = 0; kb < 2; kb++) {
    int kOff = kb * 32;
    int kk = (lane >> 4) * 8;
    v16h af0 = frag16(&pr[(lane & 15)][kOff + kk]);
    v16h af1 = frag16(&pr[16 + (lane & 15)][kOff + kk]);
    for (int n = 0; n < 4; n++) {
      int col = n * 16 + (lane & 15);
      v16h bf = frag16(&bvT[col][kOff + kk]);
      o[0][n] = wmma_f16(af0, bf, o[0][n]);
      o[1][n] = wmma_f16(af1, bf, o[1][n]);
    }
  }
  for (int m = 0; m < 2; m++)
    for (int n = 0; n < 4; n++)
      for (int e = 0; e < 8; e++) {
        int row = m * 16 + e + ((lane & 16) ? 8 : 0);
        int col = n * 16 + (lane & 15);
        osort[(base + r0 + row) * 64 + col] = o[m][n][e];
      }
}
// Unsort + softmax-combine over hashes, merging heads into (b, t, 256)
__global__ void k_combine(const float* __restrict__ osort, const float* __restrict__ lse,
                          const int* __restrict__ undo, float* __restrict__ out,
                          int T, int t) {
  int idx = blockIdx.x * blockDim.x + threadIdx.x;
  if (idx >= 32 * t) return;
  int bh = idx / t, i = idx % t;
  int b = bh >> 2, hh = bh & 3;
  size_t base = (size_t)bh * (size_t)(NHASH * T);
  int jdx[NHASH]; float w[NHASH]; float mx = -3.0e38f;
  for (int h = 0; h < NHASH; h++) {
    jdx[h] = undo[base + (size_t)h * T + i];
    w[h] = lse[base + jdx[h]];
    mx = fmaxf(mx, w[h]);
  }
  float s = 0.f;
  for (int h = 0; h < NHASH; h++) { w[h] = expf(w[h] - mx); s += w[h]; }
  float inv = 1.f / s;
  float* op = out + ((size_t)b * t + i) * 256 + hh * 64;
  for (int e = 0; e < 64; e++) {
    float acc = 0.f;
    for (int h = 0; h < NHASH; h++) acc += osort[(base + jdx[h]) * 64 + e] * w[h];
    op[e] = acc * inv;
  }
}
// ---------------------------------------------------------------------------
// GBST kernels
// ---------------------------------------------------------------------------
__global__ void k_gbst_conv(const int* __restrict__ src, const float* __restrict__ tok,
                            const float* __restrict__ cw, const float* __restrict__ cb,
                            float* __restrict__ out) {
  size_t i = blockIdx.x * (size_t)blockDim.x + threadIdx.x;
  size_t tot = (size_t)NB_B * 2048 * 128;
  if (i >= tot) return;
  int d = i % 128;
  size_t r = i / 128;
  int n = (int)(r % 2048);
  int b = (int)(r / 2048);
  float acc = cb[d];
  for (int k = 0; k < 4; k++) {
    int ii = n + k;
    if (ii < 2048) acc += tok[(size_t)src[b * 2048 + ii] * 128 + d] * cw[d * 4 + k];
  }
  out[i] = acc;
}
__global__ void k_gbst_br(const float* __restrict__ x, float* __restrict__ br) {
  size_t i = blockIdx.x * (size_t)blockDim.x + threadIdx.x;
  size_t tot = (size_t)NB_B * 2048 * 128;
  if (i >= tot) return;
  int d = i % 128;
  size_t r = i / 128;
  int n = (int)(r % 2048);
  int b = (int)(r / 2048);
  for (int c = 0; c < 4; c++) {
    int bs = c + 1;
    int st = (n / bs) * bs;
    float s = 0.f;
    for (int j = 0; j < bs; j++) {
      int ii = st + j;
      if (ii < 2048) s += x[((size_t)b * 2048 + ii) * 128 + d];
    }
    br[(((size_t)b * 2048 + n) * 4 + c) * 128 + d] = s / (float)bs;
  }
}
__global__ void k_gbst_score(const float* __restrict__ br, const float* __restrict__ sw,
                             const float* __restrict__ sb, float* __restrict__ sc) {
  int i = blockIdx.x * blockDim.x + threadIdx.x;
  if (i >= NB_B * 2048) return;
  float s[4];
  float mx = -3.0e38f;
  for (int c = 0; c < 4; c++) {
    const float* bp = br + ((size_t)i * 4 + c) * 128;
    float a = sb[0];
    for (int d = 0; d < 128; d++) a += bp[d] * sw[d];
    s[c] = a;
    mx = fmaxf(mx, a);
  }
  float ss = 0.f;
  for (int c = 0; c < 4; c++) { s[c] = expf(s[c] - mx); ss += s[c]; }
  float inv = 1.f / ss;
  for (int c = 0; c < 4; c++) sc[(size_t)i * 4 + c] = s[c] * inv;
}
// score-consensus attention: scores' = softmax(scores @ scores^T) @ scores
__global__ __launch_bounds__(256) void k_gbst_consensus(
    const float* __restrict__ scores, float* __restrict__ out, int n) {
  __shared__ float s4[2048][4];
  int b = blockIdx.x;
  for (int i = threadIdx.x; i < n * 4; i += 256)
    s4[i >> 2][i & 3] = scores[(size_t)b * n * 4 + i];
  __syncthreads();
  int wave = threadIdx.x >> 5, lane = threadIdx.x & 31;
  int row = blockIdx.y * 8 + wave;
  if (row >= n) return;
  float s0 = s4[row][0], s1 = s4[row][1], s2 = s4[row][2], s3 = s4[row][3];
  float mx = -3.0e38f;
  for (int j = lane; j < n; j += 32) {
    float d = s0 * s4[j][0] + s1 * s4[j][1] + s2 * s4[j][2] + s3 * s4[j][3];
    mx = fmaxf(mx, d);
  }
  mx = wred_max(mx);
  float se = 0.f, a0 = 0.f, a1 = 0.f, a2 = 0.f, a3 = 0.f;
  for (int j = lane; j < n; j += 32) {
    float d = s0 * s4[j][0] + s1 * s4[j][1] + s2 * s4[j][2] + s3 * s4[j][3];
    float e = expf(d - mx);
    se += e;
    a0 += e * s4[j][0]; a1 += e * s4[j][1]; a2 += e * s4[j][2]; a3 += e * s4[j][3];
  }
  se = wred_sum(se);
  a0 = wred_sum(a0); a1 = wred_sum(a1); a2 = wred_sum(a2); a3 = wred_sum(a3);
  if (lane == 0) {
    float inv = 1.f / se;
    float* op = out + ((size_t)b * n + row) * 4;
    op[0] = a0 * inv; op[1] = a1 * inv; op[2] = a2 * inv; op[3] = a3 * inv;
  }
}
__global__ void k_gbst_final(const float* __restrict__ br, const float* __restrict__ sc,
                             float* __restrict__ xg) {
  size_t i = blockIdx.x * (size_t)blockDim.x + threadIdx.x;
  size_t tot = (size_t)NB_B * 512 * 128;
  if (i >= tot) return;
  int d = i % 128;
  size_t r = i / 128;
  int nd = (int)(r % 512);
  int b = (int)(r / 512);
  float s = 0.f;
  for (int g = 0; g < 4; g++) {
    int n = nd * 4 + g;
    const float* bp = br + ((size_t)b * 2048 + n) * 4 * 128;
    const float* sp = sc + ((size_t)b * 2048 + n) * 4;
    for (int c = 0; c < 4; c++) s += bp[c * 128 + d] * sp[c];
  }
  xg[i] = s * 0.25f;
}
__global__ void k_dec_embed(const int* __restrict__ tgt, const float* __restrict__ tok,
                            const float* __restrict__ pos1, const float* __restrict__ pos2,
                            float* __restrict__ out) {
  size_t i = blockIdx.x * (size_t)blockDim.x + threadIdx.x;
  size_t tot = (size_t)NB_B * 256 * 256;
  if (i >= tot) return;
  int d = i % 256;
  size_t r = i / 256;
  int t = (int)(r % 256);
  int b = (int)(r / 256);
  out[i] = tok[(size_t)tgt[b * 256 + t] * 256 + d] +
           pos1[(size_t)(t >> 5) * 256 + d] + pos2[(size_t)(t & 31) * 256 + d];
}
// Normalized tied-embedding head, transposed and zero-padded to NHEAD_PAD cols
__global__ void k_build_wtT(const float* __restrict__ tok, half_t* __restrict__ out) {
  int n = blockIdx.x * blockDim.x + threadIdx.x;
  if (n >= NHEAD_PAD) return;
  if (n < 258) {
    float s = 0.f;
    for (int k = 0; k < 256; k++) { float v = tok[(size_t)n * 256 + k]; s += v * v; }
    float r = rsqrtf(fmaxf(s, 1e-24f));
    for (int k = 0; k < 256; k++)
      out[(size_t)k * NHEAD_PAD + n] = (half_t)(tok[(size_t)n * 256 + k] * r);
  } else {
    for (int k = 0; k < 256; k++) out[(size_t)k * NHEAD_PAD + n] = (half_t)0;
  }
}
// Slice first 258 of NHEAD_PAD columns into the output buffer
__global__ void k_slice_head(const float* __restrict__ in, float* __restrict__ out,
                             int rows) {
  size_t i = blockIdx.x * (size_t)blockDim.x + threadIdx.x;
  size_t tot = (size_t)rows * 258;
  if (i >= tot) return;
  size_t r = i / 258;
  int c = (int)(i % 258);
  out[i] = in[r * NHEAD_PAD + c];
}

// ---------------------------------------------------------------------------
// Host orchestration
// ---------------------------------------------------------------------------
static void gemm_launch(hipStream_t s, const half_t* A, const half_t* B,
                        const float* bias, const float* resid, float* C,
                        int M, int N, int K, int relu) {
  dim3 g((N + 63) / 64, (M + 63) / 64), b(256);
  k_gemm<<<g, b, 0, s>>>(A, B, bias, resid, C, M, N, K, relu);
}

struct LayerP {
  const float *ln1_g, *ln1_b, *Wqk, *Wv, *Wo, *bo, *ln2_g, *ln2_b, *W1, *b1, *W2, *b2;
};
struct LayerH { half_t *Wqk, *Wv, *Wo, *W1, *W2; };

extern "C" void kernel_launch(void* const* d_in, const int* in_sizes, int n_in,
                              void* d_out, int out_size, void* d_ws, size_t ws_size,
                              hipStream_t stream) {
  (void)in_sizes; (void)n_in; (void)out_size; (void)ws_size;
  // ---- inputs (pytree leaves, declaration order) ----
  int ci = 0;
  const int* src = (const int*)d_in[ci++];
  const int* tgt = (const int*)d_in[ci++];
  const float* g_tok = (const float*)d_in[ci++];
  const float* g_cw  = (const float*)d_in[ci++];
  const float* g_cb  = (const float*)d_in[ci++];
  const float* g_sw  = (const float*)d_in[ci++];
  const float* g_sb  = (const float*)d_in[ci++];
  const float* Wg    = (const float*)d_in[ci++];
  const float* bg    = (const float*)d_in[ci++];
  const float* encpe = (const float*)d_in[ci++];
  LayerP enc[12], dec[4];
  auto rdLayer = [&](LayerP& L) {
    L.ln1_g = (const float*)d_in[ci++]; L.ln1_b = (const float*)d_in[ci++];
    L.Wqk = (const float*)d_in[ci++];   L.Wv = (const float*)d_in[ci++];
    L.Wo = (const float*)d_in[ci++];    L.bo = (const float*)d_in[ci++];
    L.ln2_g = (const float*)d_in[ci++]; L.ln2_b = (const float*)d_in[ci++];
    L.W1 = (const float*)d_in[ci++];    L.b1 = (const float*)d_in[ci++];
    L.W2 = (const float*)d_in[ci++];    L.b2 = (const float*)d_in[ci++];
  };
  for (int i = 0; i < 12; i++) rdLayer(enc[i]);
  const float* dtok = (const float*)d_in[ci++];
  const float* pos1 = (const float*)d_in[ci++];
  const float* pos2 = (const float*)d_in[ci++];
  for (int i = 0; i < 4; i++) rdLayer(dec[i]);
  const float* lnf_g = (const float*)d_in[ci++];
  const float* lnf_b = (const float*)d_in[ci++];

  // ---- workspace bump allocator ----
  char* wsb = (char*)d_ws;
  size_t off = 0;
  auto alloc = [&](size_t bytes) {
    off = (off + 255) & ~(size_t)255;
    char* p = wsb + off;
    off += bytes;
    return p;
  };
  auto allocF = [&](size_t n) { return (float*)alloc(n * 4); };
  auto allocH = [&](size_t n) { return (half_t*)alloc(n * 2); };
  auto allocI = [&](size_t n) { return (int*)alloc(n * 4); };
  auto cvt = [&](const float* s, half_t* d, size_t n) {
    k_cvt_h<<<(unsigned)((n + 255) / 256), 256, 0, stream>>>(s, d, n);
  };

  // ---- f16 weights ----
  LayerH ench[12], dech[4];
  auto mkLayerH = [&](const LayerP& P, LayerH& H) {
    H.Wqk = allocH(256 * 256); cvt(P.Wqk, H.Wqk, 256 * 256);
    H.Wv  = allocH(256 * 64);  cvt(P.Wv,  H.Wv,  256 * 64);
    H.Wo  = allocH(256 * 256); cvt(P.Wo,  H.Wo,  256 * 256);
    H.W1  = allocH(256 * 1024); cvt(P.W1, H.W1, 256 * 1024);
    H.W2  = allocH(1024 * 256); cvt(P.W2, H.W2, 1024 * 256);
  };
  for (int i = 0; i < 12; i++) mkLayerH(enc[i], ench[i]);
  for (int i = 0; i < 4; i++)  mkLayerH(dec[i], dech[i]);
  half_t* Wg_h  = allocH(128 * 256); cvt(Wg, Wg_h, 128 * 256);
  half_t* wtT_h = allocH(256 * NHEAD_PAD);
  k_build_wtT<<<(NHEAD_PAD + 63) / 64, 64, 0, stream>>>(dtok, wtT_h);

  // ---- scratch (sized for the worst case: decoder T=768) ----
  const int TMAX = 768, LMAX = NHASH * TMAX, ROWSMAX = NB_B * TMAX;
  float*  xa_f  = allocF((size_t)ROWSMAX * 256);
  half_t* xa_h  = allocH((size_t)ROWSMAX * 256);
  float*  qk_f  = allocF((size_t)ROWSMAX * 256);
  float*  v_f   = allocF((size_t)ROWSMAX * 64);
  half_t* qk_h  = allocH((size_t)32 * TMAX * 64);
  half_t* v_h   = allocH((size_t)32 * TMAX * 64);
  float*  rot   = allocF(64 * 8 * 12);
  int*    keysB = allocI((size_t)32 * LMAX);
  int*    stS   = allocI((size_t)32 * LMAX);
  int*    undoS = allocI((size_t)32 * LMAX);
  int*    stq   = allocI((size_t)32 * LMAX);
  half_t* sqk   = allocH((size_t)32 * LMAX * 64);
  half_t* sv    = allocH((size_t)32 * LMAX * 64);
  float*  osort = allocF((size_t)32 * LMAX * 64);
  float*  lseS  = allocF((size_t)32 * LMAX);
  float*  ao    = allocF((size_t)NB_B * 512 * 256);
  half_t* ao_h  = allocH((size_t)NB_B * 512 * 256);
  float*  ff1   = allocF((size_t)NB_B * 512 * 1024);
  half_t* ff1_h = allocH((size_t)NB_B * 512 * 1024);
  float*  t0    = allocF((size_t)NB_B * 512 * 256);
  half_t* t0_h  = allocH((size_t)NB_B * 512 * 256);
  float*  sA = allocF((size_t)NB_B * 512 * 256);
  float*  sB = allocF((size_t)NB_B * 512 * 256);
  float*  sC = allocF((size_t)NB_B * 512 * 256);
  float*  sD = allocF((size_t)NB_B * 512 * 256);
  float*  mem = allocF((size_t)NB_B * 512 * 256);
  float*  xconv = allocF((size_t)NB_B * 2048 * 128);
  float*  brf   = allocF((size_t)NB_B * 2048 * 4 * 128);
  float*  sc1   = allocF((size_t)NB_B * 2048 * 4);
  float*  sc2   = allocF((size_t)NB_B * 2048 * 4);
  float*  xg    = allocF((size_t)NB_B * 512 * 128);
  half_t* xg_h  = allocH((size_t)NB_B * 512 * 128);
  float*  headp = allocF((size_t)NB_B * 256 * NHEAD_PAD);
  float*  tdm_sink = allocF(64);

  // TDM/cluster probe kernel (stages first weight tile with tensor DMA)
  k_tdm_stage<<<1, 32, 0, stream>>>(g_tok, tdm_sink);

  // ---- LSH attention block ----
  auto attention = [&](const float* xln, const float* memk, const LayerP& P,
                       const LayerH& PH, int t, int causal, int seed,
                       const float* resid_x1, float* y1_out) {
    int tm = memk ? 512 : 0;
    int T = t + tm;
    int rows = NB_B * T;
    int nb = T / BUCKET;
    int nb2 = nb / 2;
    int L = NHASH * T;
    int nch = NHASH * nb;
    {
      size_t n = (size_t)rows * 256;
      k_concat_rows<<<(unsigned)((n + 255) / 256), 256, 0, stream>>>(
          xln, memk, xa_f, NB_B, t, tm, 256);
      cvt(xa_f, xa_h, n);
    }
    gemm_launch(stream, xa_h, PH.Wqk, nullptr, nullptr, qk_f, rows, 256, 256, 0);
    gemm_launch(stream, xa_h, PH.Wv,  nullptr, nullptr, v_f,  rows, 64,  256, 0);
    { size_t n = (size_t)rows * 256;
      k_split_qk<<<(unsigned)((n + 255) / 256), 256, 0, stream>>>(qk_f, qk_h, NB_B, T); }
    { size_t n = (size_t)NB_B * NHEAD * T * 64;
      k_tile_v<<<(unsigned)((n + 255) / 256), 256, 0, stream>>>(v_f, v_h, NB_B, T); }
    { int n = 64 * NHASH * nb2;
      k_gen_rot<<<(n + 255) / 256, 256, 0, stream>>>(
          rot, n, (unsigned)(seed * 2654435761u + 7u)); }
    k_buckets<<<32 * T / 32, 32, 0, stream>>>(qk_h, rot, keysB, T, nb2, L);
    { int P2 = 1; while (P2 < L) P2 <<= 1;
      k_sort<<<32, 256, (size_t)P2 * 8, stream>>>(keysB, stS, undoS, L, P2); }
    { int n = 32 * L;
      k_gather_sorted<<<(n + 255) / 256, 256, 0, stream>>>(
          qk_h, v_h, stS, sqk, sv, stq, T, L); }
    k_attn_chunk<<<32 * nch, 32, 0, stream>>>(sqk, sv, stq, osort, lseS, L, nch,
                                              causal, t);
    { int n = 32 * t;
      k_combine<<<(n + 63) / 64, 64, 0, stream>>>(osort, lseS, undoS, ao, T, t); }
    cvt(ao, ao_h, (size_t)NB_B * t * 256);
    gemm_launch(stream, ao_h, PH.Wo, P.bo, resid_x1, y1_out, NB_B * t, 256, 256, 0);
  };

  // ---- reversible two-stream reformer stack ----
  auto reformer = [&](const float* x_in, const LayerP* Ls, const LayerH* Lh, int nl,
                      const float* memk, int t, int causal, int seedBase,
                      float* outAvg) {
    int rows = NB_B * t;
    size_t n = (size_t)rows * 256;
    float* x1 = sA; float* x2 = sB; float* y1 = sC; float* x2n = sD;
    k_copy<<<(unsigned)((n + 255) / 256), 256, 0, stream>>>(x_in, x1, n);
    k_copy<<<(unsigned)((n + 255) / 256), 256, 0, stream>>>(x_in, x2, n);
    for (int i = 0; i < nl; i++) {
      k_layernorm<<<(rows + 7) / 8, 256, 0, stream>>>(x2, Ls[i].ln1_g, Ls[i].ln1_b,
                                                      t0, rows);
      attention(t0, memk, Ls[i], Lh[i], t, causal, seedBase + i, x1, y1);
      k_layernorm<<<(rows + 7) / 8, 256, 0, stream>>>(y1, Ls[i].ln2_g, Ls[i].ln2_b,
                                                      t0, rows);
      cvt(t0, t0_h, n);
      gemm_launch(stream, t0_h, Lh[i].W1, Ls[i].b1, nullptr, ff1, rows, 1024, 256, 1);
      cvt(ff1, ff1_h, (size_t)rows * 1024);
      gemm_launch(stream, ff1_h, Lh[i].W2, Ls[i].b2, x2, x2n, rows, 256, 1024, 0);
      float* o1 = x1; float* o2 = x2;
      x1 = y1; x2 = x2n; y1 = o1; x2n = o2;
    }
    k_avg2<<<(unsigned)((n + 255) / 256), 256, 0, stream>>>(x1, x2, outAvg, n);
  };

  // ---- GBST front-end ----
  { size_t n = (size_t)NB_B * 2048 * 128;
    k_gbst_conv<<<(unsigned)((n + 255) / 256), 256, 0, stream>>>(src, g_tok, g_cw,
                                                                 g_cb, xconv);
    k_gbst_br<<<(unsigned)((n + 255) / 256), 256, 0, stream>>>(xconv, brf); }
  { int n = NB_B * 2048;
    k_gbst_score<<<(n + 255) / 256, 256, 0, stream>>>(brf, g_sw, g_sb, sc1); }
  { dim3 g(NB_B, 2048 / 8);
    k_gbst_consensus<<<g, 256, 0, stream>>>(sc1, sc2, 2048); }
  { size_t n = (size_t)NB_B * 512 * 128;
    k_gbst_final<<<(unsigned)((n + 255) / 256), 256, 0, stream>>>(brf, sc2, xg);
    cvt(xg, xg_h, n); }
  gemm_launch(stream, xg_h, Wg_h, bg, nullptr, ao, NB_B * 512, 256, 128, 0);
  { size_t n = (size_t)NB_B * 512 * 256;
    k_add_pe<<<(unsigned)((n + 255) / 256), 256, 0, stream>>>(ao, encpe, NB_B, 512); }

  // ---- encoder ----
  reformer(ao, enc, ench, 12, nullptr, 512, 0, 0, mem);

  // ---- decoder ----
  { size_t n = (size_t)NB_B * 256 * 256;
    k_dec_embed<<<(unsigned)((n + 255) / 256), 256, 0, stream>>>(tgt, dtok, pos1,
                                                                 pos2, ao); }
  reformer(ao, dec, dech, 4, mem, 256, 1, 100, ff1);   // ff1 reused as (2048,256)

  // ---- final LN + tied-embedding head (padded N, then slice) ----
  { int rows = NB_B * 256;
    k_layernorm<<<(rows + 7) / 8, 256, 0, stream>>>(ff1, lnf_g, lnf_b, t0, rows);
    cvt(t0, t0_h, (size_t)rows * 256);
    gemm_launch(stream, t0_h, wtT_h, nullptr, nullptr, headp,
                rows, NHEAD_PAD, 256, 0);
    size_t n = (size_t)rows * 258;
    k_slice_head<<<(unsigned)((n + 255) / 256), 256, 0, stream>>>(
        headp, (float*)d_out, rows); }
}
